// InfraFFN_Block_23098334118188
// MI455X (gfx1250) — compile-verified, matrix operand order checked by reading, not run on hardware
//
#include <hip/hip_runtime.h>
#include <math.h>

// ---------------------------------------------------------------------------
// Types for CDNA5 WMMA (wave32, 16x16x32 bf16 -> f32)
// ---------------------------------------------------------------------------
typedef __bf16 bf16_t;
typedef bf16_t v16bf __attribute__((ext_vector_type(16)));
typedef float  v8f   __attribute__((ext_vector_type(8)));
typedef unsigned int u32x4 __attribute__((ext_vector_type(4)));

#define DIMC   192
#define HEADS  6
#define HD     32
#define NWIN   64          // tokens per 8x8 window
#define HWDIM  128
#define BATCH  8
#define MTOK   (BATCH*HWDIM*HWDIM)   // 131072 tokens / pixels
#define EPSV   1e-5f
#define ATT_SCALE 0.17677669529663687f   // 32^-0.5

union FragBF { v16bf v; u32x4 q[2]; };

__device__ __forceinline__ v16bf load_frag(const bf16_t* base, int stride, int row, int k0, int lane) {
  // 16-bit A/B fragment: lane l holds row (l&15); K chunks [kb..kb+7] and [kb+16..kb+23], kb = (l>=16)?8:0
  const bf16_t* p = base + (long)(row + (lane & 15)) * stride + (k0 + ((lane >> 4) << 3));
  FragBF f;
  f.q[0] = *(const u32x4*)(p);
  f.q[1] = *(const u32x4*)(p + 16);
  return f.v;
}

__device__ __forceinline__ v8f wmma_bf16(v16bf a, v16bf b, v8f c) {
  return __builtin_amdgcn_wmma_f32_16x16x32_bf16(false, a, false, b, (short)0, c, false, false);
}

__device__ __forceinline__ float gelu_f(float x) {
  return 0.5f * x * (1.0f + erff(x * 0.70710678118654752f));
}
__device__ __forceinline__ float sigmoid_f(float x) { return 1.0f / (1.0f + expf(-x)); }

// window-token index -> pixel index and back.  mw = ((b*16+hw)*16+ww)*64 + i0*8+i1
__device__ __forceinline__ int w2p(int mw) {
  int n = mw & 63, b_ = mw >> 6;
  int i1 = n & 7, i0 = n >> 3;
  int ww = b_ & 15, hw = (b_ >> 4) & 15, b = b_ >> 8;
  return (((b << 7) + hw * 8 + i0) << 7) + ww * 8 + i1;
}
__device__ __forceinline__ int p2w(int mp) {
  int w = mp & 127, h = (mp >> 7) & 127, b = mp >> 14;
  return ((((b * 16 + (h >> 3)) * 16 + (w >> 3)) << 6) | ((h & 7) * 8 + (w & 7)));
}

// ---------------------------------------------------------------------------
// GEMM with fused epilogues.  out[m,o] = sum_k A[m,k]*W[o,k]  (A: MxK, W: OxK)
// ---------------------------------------------------------------------------
enum { EPI_F32 = 0, EPI_BF16, EPI_VT, EPI_BN_F32, EPI_BN_GELU_BF16,
       EPI_GELU_BF16, EPI_ADD_F32, EPI_ADD_NCHW };

struct Epi {
  int mode;
  float scale;
  const float* bias;
  const float* bn_g; const float* bn_b; const float* bn_m; const float* bn_v;
  const float* add;     // residual (MxO, row major)
  const float* gate;    // (8,192) sigmoid(ci) for V gating
  float* outF;
  bf16_t* outB;
  int oStride;
  int oOff;
};

__device__ __forceinline__ void epi_store(const Epi& e, int m, int o, float acc) {
  float v = acc + (e.bias ? e.bias[o] : 0.0f);
  switch (e.mode) {
    case EPI_F32:
      e.outF[(long)m * e.oStride + e.oOff + o] = v; break;
    case EPI_BF16:
      e.outB[(long)m * e.oStride + e.oOff + o] = (bf16_t)(v * e.scale); break;
    case EPI_VT: {  // gated V, transposed per (window,head): vt[(b_*H+h)*32+d][n]
      int b_ = m >> 6, n = m & 63, b = b_ >> 8;
      v *= e.gate[b * DIMC + o];
      int h = o >> 5, d = o & 31;
      e.outB[((long)(b_ * HEADS + h) * HD + d) * NWIN + n] = (bf16_t)v;
    } break;
    case EPI_BN_F32: {
      float s = e.bn_g[o] * rsqrtf(e.bn_v[o] + EPSV);
      e.outF[(long)m * e.oStride + e.oOff + o] = v * s + e.bn_b[o] - e.bn_m[o] * s;
    } break;
    case EPI_BN_GELU_BF16: {
      float s = e.bn_g[o] * rsqrtf(e.bn_v[o] + EPSV);
      e.outB[(long)m * e.oStride + e.oOff + o] = (bf16_t)gelu_f(v * s + e.bn_b[o] - e.bn_m[o] * s);
    } break;
    case EPI_GELU_BF16:
      e.outB[(long)m * e.oStride + o] = (bf16_t)gelu_f(v); break;
    case EPI_ADD_F32:
      e.outF[(long)m * e.oStride + o] = v + e.add[(long)m * e.oStride + o]; break;
    case EPI_ADD_NCHW: {  // final residual + window-reverse into NCHW output
      float r = v + e.add[(long)m * DIMC + o];
      int pix = w2p(m); int b = pix >> 14, sp = pix & 16383;
      e.outF[((long)(b * DIMC + o) << 14) + sp] = r;
    } break;
  }
}

// NJ is compile-time so acc[] stays in statically-indexed VGPRs (no v_movrel).
template <int NJ>
__global__ void __launch_bounds__(256)
k_gemm(const bf16_t* __restrict__ A, const bf16_t* __restrict__ W,
       int M, int K, int O, Epi e) {
  int lane = threadIdx.x & 31;
  int wid  = blockIdx.x * (blockDim.x >> 5) + (threadIdx.x >> 5);
  int oT = O / (16 * NJ);
  int tiles = (M >> 4) * oT;
  if (wid >= tiles) return;                 // wave-uniform
  int ot = wid % oT, mt = wid / oT;
  const bf16_t* Ab = A + (long)mt * 16 * K;
  const bf16_t* Wb = W + (long)ot * 16 * NJ * K;
  v8f z = {0.f,0.f,0.f,0.f,0.f,0.f,0.f,0.f};
  v8f acc[NJ];
#pragma unroll
  for (int j = 0; j < NJ; ++j) acc[j] = z;
  for (int k0 = 0; k0 < K; k0 += 32) {
    v16bf a = load_frag(Ab, K, 0, k0, lane);
#pragma unroll
    for (int j = 0; j < NJ; ++j) {
      v16bf b = load_frag(Wb, K, j * 16, k0, lane);
      acc[j] = wmma_bf16(a, b, acc[j]);
    }
  }
  int m0 = mt * 16 + ((lane >> 4) << 3);    // C/D layout: M = r + 8*(lane>=16), N = lane&15
  int o0 = ot * 16 * NJ + (lane & 15);
#pragma unroll
  for (int j = 0; j < NJ; ++j)
#pragma unroll
    for (int r = 0; r < 8; ++r)
      epi_store(e, m0 + r, o0 + j * 16, acc[j][r]);
}

// ---------------------------------------------------------------------------
// Windowed attention: one workgroup (4 waves) per (window, head)
// ---------------------------------------------------------------------------
__global__ void __launch_bounds__(128)
k_attn(const bf16_t* __restrict__ q, const bf16_t* __restrict__ kk,
       const bf16_t* __restrict__ vt, const float* __restrict__ rel_tab,
       float* __restrict__ xaF, bf16_t* __restrict__ xaPB) {
  __shared__ float  S[NWIN * NWIN];
  __shared__ __align__(16) bf16_t P[NWIN * NWIN];
  int wh = blockIdx.x;
  int b_ = wh / HEADS, h = wh % HEADS;
  int lane = threadIdx.x & 31, w = threadIdx.x >> 5;
  const bf16_t* qb = q  + (long)b_ * NWIN * DIMC + h * HD;
  const bf16_t* kb = kk + (long)b_ * NWIN * DIMC + h * HD;
  const bf16_t* vb = vt + (long)(b_ * HEADS + h) * HD * NWIN;

  // S = (q*scale) @ k^T  (scale already folded into q)
  v8f z = {0.f,0.f,0.f,0.f,0.f,0.f,0.f,0.f};
  v8f acc[4] = {z, z, z, z};
  v16bf a = load_frag(qb, DIMC, w * 16, 0, lane);
#pragma unroll
  for (int j = 0; j < 4; ++j) {
    v16bf bfrag = load_frag(kb, DIMC, j * 16, 0, lane);
    acc[j] = wmma_bf16(a, bfrag, acc[j]);
  }
  int hi = (lane >> 4) << 3;
#pragma unroll
  for (int j = 0; j < 4; ++j)
#pragma unroll
    for (int r = 0; r < 8; ++r)
      S[(w * 16 + r + hi) * NWIN + j * 16 + (lane & 15)] = acc[j][r];
  __syncthreads();

  // bias + softmax (one thread per row)
  if (threadIdx.x < NWIN) {
    int n = threadIdx.x, i0 = n >> 3, i1 = n & 7;
    float mx = -3.4e38f;
    for (int mc = 0; mc < NWIN; ++mc) {
      int j0 = mc >> 3, j1 = mc & 7;
      float s = S[n * NWIN + mc] + rel_tab[((i0 - j0 + 7) * 15 + (i1 - j1 + 7)) * HEADS + h];
      S[n * NWIN + mc] = s;
      mx = fmaxf(mx, s);
    }
    float sum = 0.f;
    for (int mc = 0; mc < NWIN; ++mc) { float ev = expf(S[n * NWIN + mc] - mx); S[n * NWIN + mc] = ev; sum += ev; }
    float inv = 1.f / sum;
    for (int mc = 0; mc < NWIN; ++mc) P[n * NWIN + mc] = (bf16_t)(S[n * NWIN + mc] * inv);
  }
  __syncthreads();

  // out = P @ V  (V pre-transposed: vt[d][n])
  v8f o0 = z, o1 = z;
#pragma unroll
  for (int ks = 0; ks < 2; ++ks) {
    v16bf pa = load_frag(P,  NWIN, w * 16, ks * 32, lane);
    v16bf b0 = load_frag(vb, NWIN, 0,      ks * 32, lane);
    v16bf b1 = load_frag(vb, NWIN, 16,     ks * 32, lane);
    o0 = wmma_bf16(pa, b0, o0);
    o1 = wmma_bf16(pa, b1, o1);
  }
#pragma unroll
  for (int r = 0; r < 8; ++r) {
    int tok = b_ * NWIN + w * 16 + r + hi;
    int pix = w2p(tok);
    int d0 = lane & 15;
    float v0 = o0[r], v1 = o1[r];
    xaF [(long)tok * DIMC + h * HD + d0]       = v0;
    xaF [(long)tok * DIMC + h * HD + 16 + d0]  = v1;
    xaPB[(long)pix * DIMC + h * HD + d0]       = (bf16_t)v0;
    xaPB[(long)pix * DIMC + h * HD + 16 + d0]  = (bf16_t)v1;
  }
}

// ---------------------------------------------------------------------------
// LayerNorm: one wave per row; optional layout permutation on output.
// map: 0 = identity, 1 = window->pixel, 2 = pixel->window
// ---------------------------------------------------------------------------
__global__ void __launch_bounds__(256)
k_ln(const float* __restrict__ in, const float* __restrict__ g, const float* __restrict__ b,
     int Mrows, int C, int map, float* outF, int ofS, bf16_t* outB, int obS, int obO) {
  int wid = blockIdx.x * (blockDim.x >> 5) + (threadIdx.x >> 5);
  if (wid >= Mrows) return;
  int lane = threadIdx.x & 31;
  const float* row = in + (long)wid * C;
  float s = 0.f, s2 = 0.f;
  for (int c = lane; c < C; c += 32) { float x = row[c]; s += x; s2 += x * x; }
  for (int msk = 16; msk; msk >>= 1) { s += __shfl_xor(s, msk, 32); s2 += __shfl_xor(s2, msk, 32); }
  float mean = s / C;
  float inv  = rsqrtf(s2 / C - mean * mean + EPSV);
  int mo = (map == 1) ? w2p(wid) : (map == 2) ? p2w(wid) : wid;
  for (int c = lane; c < C; c += 32) {
    float y = (row[c] - mean) * inv * g[c] + b[c];
    if (outF) outF[(long)mo * ofS + c] = y;
    if (outB) outB[(long)mo * obS + obO + c] = (bf16_t)y;
  }
}

// window-partition the NCHW input into (B_,N,C) fp32 + bf16 copies
__global__ void k_wpart(const float* __restrict__ inp, float* __restrict__ xF, bf16_t* __restrict__ xB) {
  long idx = (long)blockIdx.x * 256 + threadIdx.x;
  if (idx >= (long)MTOK * DIMC) return;
  int c  = (int)(idx % DIMC);
  int mw = (int)(idx / DIMC);
  int pix = w2p(mw);
  int b = pix >> 14, sp = pix & 16383;
  float v = inp[((long)(b * DIMC + c) << 14) + sp];
  xF[idx] = v;
  xB[idx] = (bf16_t)v;
}

// depthwise conv (SAME) + BN + GELU on pixel-major NHWC, bf16 out
__global__ void k_dwconv(const float* __restrict__ inP, const float* __restrict__ w,
                         const float* __restrict__ bias, const float* __restrict__ g,
                         const float* __restrict__ bb, const float* __restrict__ m,
                         const float* __restrict__ vv, int ks, bf16_t* __restrict__ out) {
  long idx = (long)blockIdx.x * 256 + threadIdx.x;
  if (idx >= (long)MTOK * DIMC) return;
  int c = (int)(idx % DIMC);
  long pix = idx / DIMC;
  int x = (int)(pix & 127), y = (int)((pix >> 7) & 127);
  long b = pix >> 14;
  int r = ks >> 1;
  float acc = 0.f;
  for (int dy = -r; dy <= r; ++dy)
    for (int dx = -r; dx <= r; ++dx) {
      int yy = y + dy, xx = x + dx;
      if (yy < 0 || yy > 127 || xx < 0 || xx > 127) continue;
      acc += w[(c * ks + (dy + r)) * ks + (dx + r)] *
             inP[(((b << 14) + (yy << 7) + xx) * DIMC) + c];
    }
  acc += bias[c];
  float s = g[c] * rsqrtf(vv[c] + EPSV);
  out[idx] = (bf16_t)gelu_f(acc * s + bb[c] - m[c] * s);
}

// spatial mean per (b,c) over pixels
__global__ void k_colmean(const float* __restrict__ xcnn, float* __restrict__ mean) {
  int bc = blockIdx.x; int b = bc / DIMC, c = bc % DIMC;
  float s = 0.f;
  for (int p = threadIdx.x; p < 16384; p += 256)
    s += xcnn[((long)(b * 16384 + p)) * DIMC + c];
  __shared__ float red[256];
  red[threadIdx.x] = s; __syncthreads();
  for (int st = 128; st; st >>= 1) { if ((int)threadIdx.x < st) red[threadIdx.x] += red[threadIdx.x + st]; __syncthreads(); }
  if (threadIdx.x == 0) mean[bc] = red[0] / 16384.f;
}

// channel interaction: ci2(gelu(bn(ci1(mean)))) -> sigmoid -> gate(8,192)
__global__ void k_ci(const float* __restrict__ mean,
                     const float* ci1_w, const float* ci1_b,
                     const float* g, const float* bb, const float* m, const float* vv,
                     const float* ci2_w, const float* ci2_b, float* __restrict__ gate) {
  __shared__ float mid[96];
  int b = blockIdx.x, t = threadIdx.x;
  if (t < 96) {
    float a = ci1_b[t];
    for (int c = 0; c < DIMC; ++c) a += ci1_w[t * DIMC + c] * mean[b * DIMC + c];
    float s = g[t] * rsqrtf(vv[t] + EPSV);
    mid[t] = gelu_f(a * s + bb[t] - m[t] * s);
  }
  __syncthreads();
  float a = ci2_b[t];
  for (int j = 0; j < 96; ++j) a += ci2_w[t * 96 + j] * mid[j];
  gate[b * DIMC + t] = sigmoid_f(a);
}

// spatial gate: sigmoid(si2(si1_act)) scales x_cnn in place (per pixel)
__global__ void k_spatial_gate(const bf16_t* __restrict__ si, const float* __restrict__ w2,
                               const float* __restrict__ b2, float* __restrict__ xcnn) {
  int pix = blockIdx.x * 256 + threadIdx.x;
  if (pix >= MTOK) return;
  const bf16_t* s = si + (long)pix * 96;
  float a = b2[0];
  for (int j = 0; j < 96; ++j) a += (float)s[j] * w2[j];
  float gsig = sigmoid_f(a);
  float* xr = xcnn + (long)pix * DIMC;
  for (int c = 0; c < DIMC; ++c) xr[c] *= gsig;
}

__global__ void k_f2b(const float* __restrict__ s, bf16_t* __restrict__ d, int n) {
  int i = blockIdx.x * 256 + threadIdx.x;
  if (i < n) d[i] = (bf16_t)s[i];
}

// ---------------------------------------------------------------------------
// Host orchestration
// ---------------------------------------------------------------------------
extern "C" void kernel_launch(void* const* d_in, const int* in_sizes, int n_in,
                              void* d_out, int out_size, void* d_ws, size_t ws_size,
                              hipStream_t stream) {
#define F(i) ((const float*)d_in[i])
  const float* inp = F(0);

  char* base = (char*)d_ws;
  size_t off = 0;
  auto alloc = [&](size_t bytes) { char* p = base + off; off += (bytes + 255) & ~(size_t)255; return (void*)p; };
  const size_t S192F = (size_t)MTOK * DIMC * 4;
  const size_t S192B = (size_t)MTOK * DIMC * 2;

  float*  xwF   = (float*)alloc(S192F);              // shortcut (W layout)
  bf16_t* xwB   = (bf16_t*)alloc(S192B);
  float*  g0    = (float*)alloc(S192F);              // GEMM scratch; later xa f32
  bf16_t* xalnB = (bf16_t*)alloc(S192B);
  float*  xc1P  = (float*)alloc(S192F);              // later reused as x_res
  float*  xc2P  = (float*)alloc(S192F);
  bf16_t* cnv1B = (bf16_t*)alloc(S192B);             // later reused as q
  bf16_t* cnv2B = (bf16_t*)alloc(S192B);             // later reused as k
  bf16_t* vtB   = (bf16_t*)alloc(S192B);
  float*  xcnn  = (float*)alloc(S192F);
  bf16_t* xaPB  = (bf16_t*)alloc(S192B);
  bf16_t* siB   = (bf16_t*)alloc((size_t)MTOK * 96 * 2);
  bf16_t* catI  = (bf16_t*)alloc((size_t)MTOK * 384 * 2);
  bf16_t* catO  = (bf16_t*)alloc(S192B);
  bf16_t* hB    = (bf16_t*)alloc(S192B);
  bf16_t* fc1B  = (bf16_t*)alloc((size_t)MTOK * 768 * 2);
  float*  meanB = (float*)alloc(BATCH * DIMC * 4);
  float*  gateB = (float*)alloc(BATCH * DIMC * 4);
  bf16_t* w_pa  = (bf16_t*)alloc(192 * 192 * 2);
  bf16_t* w_pc1 = (bf16_t*)alloc(192 * 192 * 2);
  bf16_t* w_pc2 = (bf16_t*)alloc(192 * 192 * 2);
  bf16_t* w_qkv = (bf16_t*)alloc(576 * 192 * 2);
  bf16_t* w_p1  = (bf16_t*)alloc(96 * 192 * 2);
  bf16_t* w_p2  = (bf16_t*)alloc(96 * 192 * 2);
  bf16_t* w_si1 = (bf16_t*)alloc(96 * 192 * 2);
  bf16_t* w_cat = (bf16_t*)alloc(192 * 384 * 2);
  bf16_t* w_pr  = (bf16_t*)alloc(192 * 192 * 2);
  bf16_t* w_fc1 = (bf16_t*)alloc(768 * 192 * 2);
  bf16_t* w_fc2 = (bf16_t*)alloc(192 * 768 * 2);

  float*  xres = xc1P;   // reuse (free after dw3)
  float*  xaF  = g0;     // reuse (free after pc2 LN)
  bf16_t* qB   = cnv1B;  // reuse (free after p1 GEMM)
  bf16_t* kB   = cnv2B;  // reuse (free after p2 GEMM)

  auto cvt = [&](const float* s, bf16_t* d, int n) {
    k_f2b<<<(n + 255) / 256, 256, 0, stream>>>(s, d, n);
  };
  cvt(F(1),  w_pa, 36864);  cvt(F(5),  w_pc1, 36864); cvt(F(9),  w_pc2, 36864);
  cvt(F(45), w_qkv, 110592); cvt(F(25), w_p1, 18432); cvt(F(31), w_p2, 18432);
  cvt(F(48), w_si1, 18432); cvt(F(60), w_cat, 73728); cvt(F(62), w_pr, 36864);
  cvt(F(66), w_fc1, 147456); cvt(F(68), w_fc2, 147456);

  auto gemm = [&](const bf16_t* A, const bf16_t* W, int M, int K, int O, int nj, Epi e) {
    int tiles = (M / 16) * (O / (16 * nj));
    int blocks = (tiles + 7) / 8;
    if (nj == 4) k_gemm<4><<<blocks, 256, 0, stream>>>(A, W, M, K, O, e);
    else         k_gemm<2><<<blocks, 256, 0, stream>>>(A, W, M, K, O, e);
  };
  auto mk = [](int mode) { Epi e{}; e.mode = mode; e.scale = 1.f; return e; };

  long tot = (long)MTOK * DIMC;
  int  totB = (int)((tot + 255) / 256);

  // 1. window partition
  k_wpart<<<totB, 256, 0, stream>>>(inp, xwF, xwB);

  // 2. pa -> LN(lna) -> bf16
  { Epi e = mk(EPI_F32); e.bias = F(2); e.outF = g0; e.oStride = DIMC;
    gemm(xwB, w_pa, MTOK, 192, 192, 4, e); }
  k_ln<<<MTOK / 8, 256, 0, stream>>>(g0, F(3), F(4), MTOK, DIMC, 0, nullptr, 0, xalnB, DIMC, 0);

  // 3. pc1 -> LN -> pixel layout f32 ; pc2 likewise
  { Epi e = mk(EPI_F32); e.bias = F(6); e.outF = g0; e.oStride = DIMC;
    gemm(xwB, w_pc1, MTOK, 192, 192, 4, e); }
  k_ln<<<MTOK / 8, 256, 0, stream>>>(g0, F(7), F(8), MTOK, DIMC, 1, xc1P, DIMC, nullptr, 0, 0);
  { Epi e = mk(EPI_F32); e.bias = F(10); e.outF = g0; e.oStride = DIMC;
    gemm(xwB, w_pc2, MTOK, 192, 192, 4, e); }
  k_ln<<<MTOK / 8, 256, 0, stream>>>(g0, F(11), F(12), MTOK, DIMC, 1, xc2P, DIMC, nullptr, 0, 0);

  // 4. conv branches: dw+BN+GELU, then 1x1 conv (GEMM) + BN into x_cnn halves
  k_dwconv<<<totB, 256, 0, stream>>>(xc1P, F(13), F(14), F(15), F(16), F(17), F(18), 3, cnv1B);
  { Epi e = mk(EPI_BN_F32); e.bias = F(26); e.bn_g = F(27); e.bn_b = F(28); e.bn_m = F(29); e.bn_v = F(30);
    e.outF = xcnn; e.oStride = DIMC; e.oOff = 0;
    gemm(cnv1B, w_p1, MTOK, 192, 96, 2, e); }
  k_dwconv<<<totB, 256, 0, stream>>>(xc2P, F(19), F(20), F(21), F(22), F(23), F(24), 5, cnv2B);
  { Epi e = mk(EPI_BN_F32); e.bias = F(32); e.bn_g = F(33); e.bn_b = F(34); e.bn_m = F(35); e.bn_v = F(36);
    e.outF = xcnn; e.oStride = DIMC; e.oOff = 96;
    gemm(cnv2B, w_p2, MTOK, 192, 96, 2, e); }

  // 5. channel-interaction gate
  k_colmean<<<BATCH * DIMC, 256, 0, stream>>>(xcnn, meanB);
  k_ci<<<BATCH, 192, 0, stream>>>(meanB, F(37), F(38), F(39), F(40), F(41), F(42), F(43), F(44), gateB);

  // 6. qkv projections (q scaled, v gated + transposed)
  { Epi e = mk(EPI_BF16); e.scale = ATT_SCALE; e.bias = F(46); e.outB = qB; e.oStride = DIMC;
    gemm(xalnB, w_qkv, MTOK, 192, 192, 4, e); }
  { Epi e = mk(EPI_BF16); e.bias = F(46) + 192; e.outB = kB; e.oStride = DIMC;
    gemm(xalnB, w_qkv + 192 * 192, MTOK, 192, 192, 4, e); }
  { Epi e = mk(EPI_VT); e.bias = F(46) + 384; e.gate = gateB; e.outB = vtB;
    gemm(xalnB, w_qkv + 2 * 192 * 192, MTOK, 192, 192, 4, e); }

  // 7. attention
  k_attn<<<(MTOK / NWIN) * HEADS, 128, 0, stream>>>(qB, kB, vtB, F(47), xaF, xaPB);

  // 8. spatial gate on x_cnn
  { Epi e = mk(EPI_BN_GELU_BF16); e.bias = F(49); e.bn_g = F(50); e.bn_b = F(51); e.bn_m = F(52); e.bn_v = F(53);
    e.outB = siB; e.oStride = 96;
    gemm(xaPB, w_si1, MTOK, 192, 96, 2, e); }
  k_spatial_gate<<<MTOK / 256, 256, 0, stream>>>(siB, F(54), F(55), xcnn);

  // 9. LN both halves into concat buffer (window order)
  k_ln<<<MTOK / 8, 256, 0, stream>>>(xcnn, F(58), F(59), MTOK, DIMC, 2, nullptr, 0, catI, 384, 0);
  k_ln<<<MTOK / 8, 256, 0, stream>>>(xaF,  F(56), F(57), MTOK, DIMC, 0, nullptr, 0, catI, 384, 192);

  // 10. cat -> pr (+ shortcut)
  { Epi e = mk(EPI_BF16); e.bias = F(61); e.outB = catO; e.oStride = DIMC;
    gemm(catI, w_cat, MTOK, 384, 192, 4, e); }
  { Epi e = mk(EPI_ADD_F32); e.bias = F(63); e.add = xwF; e.outF = xres; e.oStride = DIMC;
    gemm(catO, w_pr, MTOK, 192, 192, 4, e); }

  // 11. MLP + residual, window-reverse into d_out (NCHW f32)
  k_ln<<<MTOK / 8, 256, 0, stream>>>(xres, F(64), F(65), MTOK, DIMC, 0, nullptr, 0, hB, DIMC, 0);
  { Epi e = mk(EPI_GELU_BF16); e.bias = F(67); e.outB = fc1B; e.oStride = 768;
    gemm(hB, w_fc1, MTOK, 192, 768, 4, e); }
  { Epi e = mk(EPI_ADD_NCHW); e.bias = F(69); e.add = xres; e.outF = (float*)d_out;
    gemm(fc1B, w_fc2, MTOK, 768, 192, 4, e); }

  (void)in_sizes; (void)n_in; (void)out_size; (void)ws_size;
#undef F
}
